// Discriminator_63909113365211
// MI455X (gfx1250) — matrix operand order repaired and checked by reading, not compiled, not run on hardware
//
#include <hip/hip_runtime.h>
#include <hip/hip_bf16.h>
#include <hip/hip_fp16.h>

typedef __attribute__((ext_vector_type(16))) _Float16 v16h;
typedef __attribute__((ext_vector_type(8)))  _Float16 v8h;
typedef __attribute__((ext_vector_type(8)))  float    v8f;

#define LDS_STRIDE 520   // 16B-aligned row stride in halves, offsets banks by 4

// ---------------------------------------------------------------- utilities
__global__ void fill0_kernel(float* p, long long n) {
    long long i = (long long)blockIdx.x * blockDim.x + threadIdx.x;
    long long step = (long long)gridDim.x * blockDim.x;
    for (; i < n; i += step) p[i] = 0.f;
}

// scatter: agg[dst[e]] += h[src[e]]   (float4 per thread, f32 atomics -> L2)
__global__ void scatter_add_kernel(const float* __restrict__ h,
                                   const long long* __restrict__ src,
                                   const long long* __restrict__ dst,
                                   float* __restrict__ agg,
                                   long long nwork, int chunks, int K) {
    long long idx = (long long)blockIdx.x * blockDim.x + threadIdx.x;
    if (idx >= nwork) return;
    long long e = idx / chunks;
    int c = (int)(idx % chunks) * 4;
    int s = (int)src[e];
    int d = (int)dst[e];
    const float4 v = *(const float4*)(h + (long long)s * K + c);
    float* a = agg + (long long)d * K + c;
    atomicAdd(a + 0, v.x);
    atomicAdd(a + 1, v.y);
    atomicAdd(a + 2, v.z);
    atomicAdd(a + 3, v.w);
}

// fp32 [K,D] row-major -> fp16 [D,K] (so B fragments are contiguous along K)
__global__ void cvt_transpose_kernel(const float* __restrict__ w,
                                     _Float16* __restrict__ wt, int K, int D) {
    int idx = blockIdx.x * blockDim.x + threadIdx.x;
    if (idx >= K * D) return;
    int k = idx / D, c = idx % D;
    wt[(long long)c * K + k] = (_Float16)w[idx];
}

// ------------------------------------------------- fused dual-GEMM (WMMA f16)
// out[16 x 64 per block] = agg@Wrel + h@Wroot + bias ; 4 waves, one 16x16 tile each
__global__ void __launch_bounds__(128)
graphconv_gemm_kernel(const float* __restrict__ Arel,   // agg  [N,K] fp32
                      const float* __restrict__ Aroot,  // h    [N,K] fp32
                      const _Float16* __restrict__ WrelT,  // [D,K] fp16
                      const _Float16* __restrict__ WrootT, // [D,K] fp16
                      const float* __restrict__ bias,
                      float* __restrict__ out,          // [N,D] fp32
                      int K, int D) {
    __shared__ _Float16 sA[2][16 * LDS_STRIDE];
    const int tid = threadIdx.x;
    const int rowBase = blockIdx.x * 16;
    const int kv = K >> 2;                     // float4s per row

    // stage both 16xK A-panels into LDS as f16 (N % 16 == 0, no guards)
    for (int i = tid; i < 16 * kv; i += 128) {
        int r  = i / kv;
        int c4 = (i % kv) << 2;
        long long off = (long long)(rowBase + r) * K + c4;
        float4 a = *(const float4*)(Arel + off);
        float4 b = *(const float4*)(Aroot + off);
        _Float16* pa = &sA[0][r * LDS_STRIDE + c4];
        _Float16* pb = &sA[1][r * LDS_STRIDE + c4];
        pa[0] = (_Float16)a.x; pa[1] = (_Float16)a.y;
        pa[2] = (_Float16)a.z; pa[3] = (_Float16)a.w;
        pb[0] = (_Float16)b.x; pb[1] = (_Float16)b.y;
        pb[2] = (_Float16)b.z; pb[3] = (_Float16)b.w;
    }
    __syncthreads();

    const int wave = tid >> 5;
    const int lane = tid & 31;
    const int half = lane >> 4;                // 0: lanes 0-15, 1: lanes 16-31
    const int n    = lane & 15;
    const int col  = blockIdx.y * 64 + wave * 16 + n;   // D % 64 == 0

    const _Float16* bRelCol  = WrelT  + (long long)col * K;
    const _Float16* bRootCol = WrootT + (long long)col * K;
    const _Float16* aRelRow  = &sA[0][n * LDS_STRIDE];  // A row M = lane&15
    const _Float16* aRootRow = &sA[1][n * LDS_STRIDE];

    v8f acc = {};
    for (int kt = 0; kt < K; kt += 32) {
        // A 16x32 frag: K = kt + 16*half + [0..15], contiguous (matches B)
        v8h alo = *(const v8h*)(aRelRow + kt + 16 * half);
        v8h ahi = *(const v8h*)(aRelRow + kt + 16 * half + 8);
        v16h af = __builtin_shufflevector(alo, ahi,
                      0,1,2,3,4,5,6,7,8,9,10,11,12,13,14,15);
        // B 32x16 frag: K = kt + 16*half + [0..15], contiguous in [D,K] layout
        v16h bf = *(const v16h*)(bRelCol + kt + 16 * half);
        acc = __builtin_amdgcn_wmma_f32_16x16x32_f16(
                  false, af, false, bf, (short)0, acc, false, false);

        v8h rlo = *(const v8h*)(aRootRow + kt + 16 * half);
        v8h rhi = *(const v8h*)(aRootRow + kt + 16 * half + 8);
        v16h rf = __builtin_shufflevector(rlo, rhi,
                      0,1,2,3,4,5,6,7,8,9,10,11,12,13,14,15);
        v16h bf2 = *(const v16h*)(bRootCol + kt + 16 * half);
        acc = __builtin_amdgcn_wmma_f32_16x16x32_f16(
                  false, rf, false, bf2, (short)0, acc, false, false);
    }

    const float bcol = bias[col];
    #pragma unroll
    for (int v = 0; v < 8; ++v) {             // C/D: M = v + 8*half, N = lane&15
        int row = rowBase + v + 8 * half;
        out[(long long)row * D + col] = acc[v] + bcol;
    }
}

// ------------------------------------------------------- LayerNorm + LeakyReLU
__global__ void ln_act_kernel(const float* __restrict__ tmp,
                              const float* __restrict__ g,
                              const float* __restrict__ beta,
                              float* __restrict__ hout, int D) {
    __shared__ float rs1[128], rs2[128];
    const int row = blockIdx.x;
    const float* x = tmp + (long long)row * D;
    float s = 0.f, s2 = 0.f;
    for (int c = threadIdx.x; c < D; c += 128) {
        float v = x[c]; s += v; s2 += v * v;
    }
    rs1[threadIdx.x] = s; rs2[threadIdx.x] = s2;
    __syncthreads();
    for (int off = 64; off > 0; off >>= 1) {
        if (threadIdx.x < off) {
            rs1[threadIdx.x] += rs1[threadIdx.x + off];
            rs2[threadIdx.x] += rs2[threadIdx.x + off];
        }
        __syncthreads();
    }
    const float mu  = rs1[0] / (float)D;
    const float var = rs2[0] / (float)D - mu * mu;
    const float r   = rsqrtf(var + 1e-5f);
    float* y = hout + (long long)row * D;
    for (int c = threadIdx.x; c < D; c += 128) {
        float v = (x[c] - mu) * r * g[c] + beta[c];
        y[c] = (v >= 0.f) ? v : 0.2f * v;
    }
}

// --------------------------------------------------------------- sum pool + FC
__global__ void pool_partial_kernel(const float* __restrict__ h,
                                    float* __restrict__ pooled, int nrows) {
    const int f  = threadIdx.x;            // 256 threads; features f and f+256
    const int r0 = blockIdx.x * 128;
    const int r1 = (r0 + 128 < nrows) ? r0 + 128 : nrows;
    float a0 = 0.f, a1 = 0.f;
    for (int r = r0; r < r1; ++r) {
        const float* p = h + (long long)r * 512;
        a0 += p[f];
        a1 += p[f + 256];
    }
    atomicAdd(&pooled[f], a0);
    atomicAdd(&pooled[f + 256], a1);
}

__global__ void pool_final_kernel(const float* __restrict__ pooled,
                                  const float* __restrict__ fcW,
                                  const float* __restrict__ fcb,
                                  float* __restrict__ out) {
    __shared__ float red[512];
    const int t = threadIdx.x;
    red[t] = pooled[t] * fcW[t];
    __syncthreads();
    for (int off = 256; off > 0; off >>= 1) {
        if (t < off) red[t] += red[t + off];
        __syncthreads();
    }
    if (t == 0) out[0] = red[0] + fcb[0];
}

// ---------------------------------------------------------------------- host
extern "C" void kernel_launch(void* const* d_in, const int* in_sizes, int n_in,
                              void* d_out, int out_size, void* d_ws, size_t ws_size,
                              hipStream_t stream) {
    (void)n_in; (void)out_size; (void)ws_size;
    static const int dims[6] = {64, 64, 128, 256, 512, 512};

    const float* x = (const float*)d_in[0];
    const long long* ei = (const long long*)d_in[1];
    const int E = in_sizes[1] / 2;
    const int N = in_sizes[0] / dims[0];          // 20000
    const long long* src = ei;
    const long long* dst = ei + E;

    char* ws = (char*)d_ws;
    const size_t hbytes = (size_t)N * 512 * sizeof(float);
    float* agg  = (float*)ws; ws += hbytes;
    float* tmp  = (float*)ws; ws += hbytes;
    float* hbuf = (float*)ws; ws += hbytes;
    _Float16* wrelT  = (_Float16*)ws; ws += (size_t)512 * 512 * sizeof(_Float16);
    _Float16* wrootT = (_Float16*)ws; ws += (size_t)512 * 512 * sizeof(_Float16);
    float* pooled = (float*)ws; ws += 512 * sizeof(float);

    const float* h_in = x;
    for (int i = 0; i < 5; ++i) {
        const int K = dims[i], D = dims[i + 1];
        const float* Wrel  = (const float*)d_in[2 + 5 * i];
        const float* Wroot = (const float*)d_in[3 + 5 * i];
        const float* bias  = (const float*)d_in[4 + 5 * i];
        const float* gam   = (const float*)d_in[5 + 5 * i];
        const float* bet   = (const float*)d_in[6 + 5 * i];

        fill0_kernel<<<1024, 256, 0, stream>>>(agg, (long long)N * K);

        const int chunks = K / 4;
        const long long nwork = (long long)E * chunks;
        const int sblocks = (int)((nwork + 255) / 256);
        scatter_add_kernel<<<sblocks, 256, 0, stream>>>(h_in, src, dst, agg,
                                                        nwork, chunks, K);

        const int nw = K * D;
        cvt_transpose_kernel<<<(nw + 255) / 256, 256, 0, stream>>>(Wrel, wrelT, K, D);
        cvt_transpose_kernel<<<(nw + 255) / 256, 256, 0, stream>>>(Wroot, wrootT, K, D);

        dim3 grid(N / 16, D / 64);
        graphconv_gemm_kernel<<<grid, 128, 0, stream>>>(agg, h_in, wrelT, wrootT,
                                                        bias, tmp, K, D);

        ln_act_kernel<<<N, 128, 0, stream>>>(tmp, gam, bet, hbuf, D);
        h_in = hbuf;
    }

    fill0_kernel<<<2, 256, 0, stream>>>(pooled, 512);
    pool_partial_kernel<<<(N + 127) / 128, 256, 0, stream>>>(hbuf, pooled, N);
    pool_final_kernel<<<1, 512, 0, stream>>>(pooled,
                                             (const float*)d_in[27],
                                             (const float*)d_in[28],
                                             (float*)d_out);
}
